// TimeMoERouter_3435973837302
// MI455X (gfx1250) — matrix operand order repaired and checked by reading, not compiled, not run on hardware
//
#include <hip/hip_runtime.h>
#include <hip/hip_bf16.h>

// ---------------- problem constants ----------------
#define BB 2
#define SS 2048
#define HH 1024
#define EE 8
#define NHH 8
#define HDD 128
#define HQQ 256
#define CAPP 1536
#define MTOT (BB * SS)          // 4096
#define KPAD 1056               // 1026 padded to multiple of 32

typedef __attribute__((ext_vector_type(16))) _Float16 v16h;
typedef __attribute__((ext_vector_type(8)))  float    v8f;
typedef __attribute__((ext_vector_type(4)))  unsigned u32x4;
typedef __attribute__((ext_vector_type(8)))  unsigned u32x8;

static __device__ __forceinline__ v8f wmma_f16(v16h a, v16h b, v8f c) {
  return __builtin_amdgcn_wmma_f32_16x16x32_f16(false, a, false, b, (short)0, c, false, false);
}

// ---------------- TDM descriptor builders (CDNA5 ISA) ----------------
// Group 0: [1:0]=count=1, [63:32]=lds_addr, [120:64]=global_addr, [127:126]=type=2
static __device__ __forceinline__ u32x4 tdm_g0(unsigned lds_addr, unsigned long long gaddr) {
  u32x4 g;
  g[0] = 0x1u;
  g[1] = lds_addr;
  g[2] = (unsigned)gaddr;
  g[3] = (unsigned)((gaddr >> 32) & 0x1FFFFFFull) | (2u << 30);
  return g;
}
// Group 1: data_size=2B; tensor_dim0/1, tile_dim0/1, tensor_dim0_stride; no pad/iterate/multicast
static __device__ __forceinline__ u32x8 tdm_g1(unsigned td0, unsigned td1,
                                               unsigned tile0, unsigned tile1,
                                               unsigned long long stride0) {
  u32x8 g;
  g[0] = 0x00010000u;                                   // [17:16] data_size = 1 (2 bytes)
  g[1] = (td0 & 0xFFFFu) << 16;                         // tensor_dim0[15:0] @ bit48
  g[2] = (td0 >> 16) | ((td1 & 0xFFFFu) << 16);         // tensor_dim0[31:16] | tensor_dim1[15:0]
  g[3] = (td1 >> 16) | (tile0 << 16);                   // tensor_dim1[31:16] | tile_dim0
  g[4] = tile1;                                         // tile_dim1 | tile_dim2=0
  g[5] = (unsigned)stride0;                             // tensor_dim0_stride[31:0]
  g[6] = (unsigned)((stride0 >> 32) & 0xFFFFull);       // stride[47:32] | dim1_stride=0
  g[7] = 0;
  return g;
}

static __device__ __forceinline__ void tdm_load(u32x4 g0, u32x8 g1) {
  u32x4 gz; gz[0] = 0; gz[1] = 0; gz[2] = 0; gz[3] = 0;
  asm volatile("tensor_load_to_lds %0, %1, %2, %3"
               :: "s"(g0), "s"(g1), "s"(gz), "s"(gz) : "memory");
}

// ---------------- workspace layout (bytes) ----------------
constexpr size_t WS_W16    = 0;
constexpr size_t W16_ELEMS = (size_t)HH * HH;
constexpr size_t WS_WE1P   = WS_W16 + 6ull * W16_ELEMS * 2;
constexpr size_t WS_SEAS   = WS_WE1P + (size_t)KPAD * HH * 2;
constexpr size_t WS_PSUM   = WS_SEAS + (size_t)SS * HQQ * 4;
constexpr size_t WS_ACT    = (WS_PSUM + 512) & ~(size_t)255;
constexpr size_t ACT_STRIDE = (size_t)MTOT * KPAD * 2;

constexpr size_t DISP_ELEMS = (size_t)BB * SS * EE * CAPP;   // 50,331,648

// ---------------- utility kernels ----------------
__global__ void zero_out_kernel(float4* __restrict__ dst, size_t n4,
                                float* __restrict__ psum) {
  size_t i = (size_t)blockIdx.x * blockDim.x + threadIdx.x;
  size_t stride = (size_t)gridDim.x * blockDim.x;
  float4 z; z.x = 0.f; z.y = 0.f; z.z = 0.f; z.w = 0.f;
  for (; i < n4; i += stride) dst[i] = z;
  if (blockIdx.x == 0 && threadIdx.x < EE) psum[threadIdx.x] = 0.f;
}

__global__ void cast_f16_kernel(const float* __restrict__ src,
                                _Float16* __restrict__ dst, int n) {
  int i = blockIdx.x * blockDim.x + threadIdx.x;
  if (i < n) dst[i] = (_Float16)src[i];
}

__global__ void pack_we1_kernel(const float* __restrict__ src,
                                _Float16* __restrict__ dst) {
  int i = blockIdx.x * blockDim.x + threadIdx.x;
  int n = KPAD * HH;
  if (i >= n) return;
  int k = i / HH, c = i % HH;
  float v = (k < HH + 2) ? src[(size_t)k * HH + c] : 0.f;
  dst[i] = (_Float16)v;
}

__global__ void build_a0_kernel(const float* __restrict__ hs,
                                _Float16* __restrict__ A0) {
  size_t i = (size_t)blockIdx.x * blockDim.x + threadIdx.x;
  size_t n = (size_t)MTOT * KPAD;
  if (i >= n) return;
  int m = (int)(i / KPAD), c = (int)(i % KPAD);
  int s = m & (SS - 1);
  float v;
  if (c < HH)            v = hs[(size_t)m * HH + c];
  else if (c == HH)      v = (float)s;
  else if (c == HH + 1)  v = __sinf((float)s * 0.2617993877991494f);
  else                   v = 0.f;
  A0[i] = (_Float16)v;
}

__global__ void seas_kernel(const float* __restrict__ Ws1, const float* __restrict__ bs1,
                            const float* __restrict__ Ws2, const float* __restrict__ bs2,
                            float* __restrict__ out) {
  int i = blockIdx.x * blockDim.x + threadIdx.x;
  if (i >= SS * HQQ) return;
  int s = i / HQQ, col = i % HQQ;
  float season = __sinf((float)s * 0.2617993877991494f);
  float acc = bs2[col];
  for (int j = 0; j < HQQ; ++j) {
    float h = fmaxf(season * Ws1[j] + bs1[j], 0.f);
    acc += h * Ws2[j * HQQ + col];
  }
  out[i] = acc;
}

// ---------------- WMMA GEMM, double-buffered TDM-staged tiles ----------------
// 128x128x32 block tile, 8 waves in 4x2; each wave = 32x64 (2 x 4 WMMA tiles).
#define EPI_NONE    0
#define EPI_RELU    1
#define EPI_POSSEAS 2
#define EPI_TRANSK  3   // write K projection transposed: Kt[b][h][hd][s]

__global__ __launch_bounds__(256) void gemm16_kernel(
    const _Float16* __restrict__ A, const _Float16* __restrict__ W,
    const float* __restrict__ bias, _Float16* __restrict__ C,
    int M, int N, int Ktot, int mode,
    const float* __restrict__ pos, const float* __restrict__ seas) {
  __shared__ _Float16 As[2][128][32];     // 16 KB
  __shared__ _Float16 Bs[2][32][128];     // 16 KB
  __shared__ _Float16 Cs[128][128];       // 32 KB

  const int tid = threadIdx.x;
  const int wave = tid >> 5, lane = tid & 31;
  const int bm = blockIdx.y * 128, bn = blockIdx.x * 128;
  const int wr = wave >> 1, wc = wave & 1;     // wave tile origin (wr*32, wc*64)

  v8f acc[2][4];
#pragma unroll
  for (int mt = 0; mt < 2; ++mt)
#pragma unroll
    for (int wn = 0; wn < 4; ++wn) acc[mt][wn] = (v8f){};

  const int arow = lane & 15;
  const int kb = (lane < 16) ? 0 : 8;
  const int lane15 = lane & 15;
  const int rb = (lane >= 16) ? 8 : 0;

  // loop-invariant TDM group-1 descriptors
  const u32x8 g1A = tdm_g1((unsigned)Ktot, (unsigned)M, 32u, 128u, (unsigned long long)Ktot);
  const u32x8 g1B = tdm_g1((unsigned)N, (unsigned)Ktot, 128u, 32u, (unsigned long long)N);
  unsigned ldsA[2], ldsB[2];
  ldsA[0] = (unsigned)(unsigned long long)(void*)&As[0][0][0];
  ldsA[1] = (unsigned)(unsigned long long)(void*)&As[1][0][0];
  ldsB[0] = (unsigned)(unsigned long long)(void*)&Bs[0][0][0];
  ldsB[1] = (unsigned)(unsigned long long)(void*)&Bs[1][0][0];

  const int nk = Ktot >> 5;   // K-steps of 32

  // prologue: fill buffer 0
  if (wave == 0) {
    unsigned long long gA = (unsigned long long)(A + (size_t)bm * Ktot);
    unsigned long long gB = (unsigned long long)(W + bn);
    tdm_load(tdm_g0(ldsA[0], gA), g1A);
    tdm_load(tdm_g0(ldsB[0], gB), g1B);
    __builtin_amdgcn_s_wait_tensorcnt(0);
  }
  __syncthreads();

  for (int ks = 0; ks < nk; ++ks) {
    const int cur = ks & 1;
    // kick off next tile's DMA into the other buffer; left outstanding during compute
    if (ks + 1 < nk && wave == 0) {
      int k0n = (ks + 1) << 5;
      unsigned long long gA = (unsigned long long)(A + (size_t)bm * Ktot + k0n);
      unsigned long long gB = (unsigned long long)(W + (size_t)k0n * N + bn);
      tdm_load(tdm_g0(ldsA[cur ^ 1], gA), g1A);
      tdm_load(tdm_g0(ldsB[cur ^ 1], gB), g1B);
    }

    v16h a0, a1;
#pragma unroll
    for (int e = 0; e < 8; ++e) {
      a0[e]     = As[cur][wr * 32 + arow][kb + e];
      a0[e + 8] = As[cur][wr * 32 + arow][kb + 16 + e];
      a1[e]     = As[cur][wr * 32 + 16 + arow][kb + e];
      a1[e + 8] = As[cur][wr * 32 + 16 + arow][kb + 16 + e];
    }
#pragma unroll
    for (int wn = 0; wn < 4; ++wn) {
      v16h bf;
#pragma unroll
      for (int e = 0; e < 8; ++e) {
        bf[e]     = Bs[cur][kb + e][wc * 64 + wn * 16 + lane15];
        bf[e + 8] = Bs[cur][kb + 16 + e][wc * 64 + wn * 16 + lane15];
      }
      acc[0][wn] = wmma_f16(a0, bf, acc[0][wn]);
      acc[1][wn] = wmma_f16(a1, bf, acc[1][wn]);
    }

    // ensure the prefetched buffer is complete before anyone reads it next iter
    if (wave == 0) __builtin_amdgcn_s_wait_tensorcnt(0);
    __syncthreads();
  }

  if (mode == EPI_TRANSK) {
    // direct transposed scalar stores: Kt[((b*8+h)*128+hd)*2048 + s]
#pragma unroll
    for (int mt = 0; mt < 2; ++mt)
#pragma unroll
      for (int wn = 0; wn < 4; ++wn) {
        int nc = bn + wc * 64 + wn * 16 + lane15;
        float bv = bias[nc];
        int h = nc >> 7, hd = nc & (HDD - 1);
#pragma unroll
        for (int v = 0; v < 8; ++v) {
          int row = bm + wr * 32 + mt * 16 + rb + v;
          int b = row >> 11, s = row & (SS - 1);
          C[(((size_t)(b * NHH + h)) * HDD + hd) * SS + s] = (_Float16)(acc[mt][wn][v] + bv);
        }
      }
  } else {
    // stage output tile in LDS, apply epilogue, then coalesced b128 stores
#pragma unroll
    for (int mt = 0; mt < 2; ++mt)
#pragma unroll
      for (int wn = 0; wn < 4; ++wn) {
        int nc = bn + wc * 64 + wn * 16 + lane15;
        float bv = bias[nc];
#pragma unroll
        for (int v = 0; v < 8; ++v) {
          int row = bm + wr * 32 + mt * 16 + rb + v;
          float val = acc[mt][wn][v] + bv;
          if (mode == EPI_RELU) val = fmaxf(val, 0.f);
          if (mode == EPI_POSSEAS) {
            int s = row & (SS - 1);
            val += pos[(size_t)s * HH + nc] + seas[(size_t)s * HQQ + (nc & (HQQ - 1))];
          }
          Cs[wr * 32 + mt * 16 + rb + v][wc * 64 + wn * 16 + lane15] = (_Float16)val;
        }
      }
    __syncthreads();
    const int r = tid >> 1, halfc = (tid & 1) * 64;
    _Float16* crow = C + (size_t)(bm + r) * N + bn + halfc;
#pragma unroll
    for (int c8 = 0; c8 < 64; c8 += 8)
      *(float4*)(crow + c8) = *(float4*)&Cs[r][halfc + c8];
  }
}

// ---------------- flash attention: per (b, h, 16-query tile) ----------------
// Kt layout: [b][h][hd=128][s=2048] f16 (written by EPI_TRANSK)
__global__ __launch_bounds__(256) void attn_kernel(
    const _Float16* __restrict__ Q, const _Float16* __restrict__ Kt,
    const _Float16* __restrict__ V, _Float16* __restrict__ Ctx) {
  __shared__ float    sc[16][512];
  __shared__ _Float16 Pl[16][512];
  __shared__ float rowmax[16], rowsum[16], rowscale[16];

  const int tid = threadIdx.x, wave = tid >> 5, lane = tid & 31;
  const int qt = blockIdx.x & 127;
  const int h  = (blockIdx.x >> 7) & 7;
  const int b  = blockIdx.x >> 10;
  const size_t bofs = (size_t)b * SS;
  const size_t ktbase = ((size_t)(b * NHH + h)) * HDD * SS;
  const int hofs = h * HDD;

  const int arow = lane & 15;
  const int kb = (lane < 16) ? 0 : 8;
  const int rb = (lane >= 16) ? 8 : 0;

  v16h qf[4];
#pragma unroll
  for (int kc = 0; kc < 4; ++kc) {
    const _Float16* qrow = Q + (bofs + qt * 16 + arow) * HH + hofs + kc * 32;
#pragma unroll
    for (int e = 0; e < 8; ++e) {
      qf[kc][e]     = qrow[kb + e];
      qf[kc][e + 8] = qrow[kb + 16 + e];
    }
  }
  if (tid < 16) { rowmax[tid] = -3.0e38f; rowsum[tid] = 0.f; }
  __syncthreads();

  v8f ctx = (v8f){};
  const float scale = 0.08838834764831845f;  // 1/sqrt(128)

  for (int ch = 0; ch < 4; ++ch) {
    // --- scores: contiguous Kt rows, 4 key tiles per wave ---
    for (int kt = wave; kt < 32; kt += 8) {
      v8f s8 = (v8f){};
      const int key0 = ch * 512 + kt * 16;
#pragma unroll
      for (int kc = 0; kc < 4; ++kc) {
        const _Float16* krow = Kt + ktbase + (size_t)(kc * 32) * SS + key0 + arow;
        v16h bf;
#pragma unroll
        for (int e = 0; e < 8; ++e) {
          bf[e]     = krow[(size_t)(kb + e) * SS];
          bf[e + 8] = krow[(size_t)(kb + 16 + e) * SS];
        }
        s8 = wmma_f16(qf[kc], bf, s8);
      }
      if (ch < 3) __builtin_prefetch(Kt + ktbase + (size_t)lane * SS + key0 + 512, 0, 1);
      const int n = lane & 15;
#pragma unroll
      for (int v = 0; v < 8; ++v) sc[rb + v][kt * 16 + n] = s8[v] * scale;
    }
    __syncthreads();

    // --- online softmax: wave handles rows 2w, 2w+1 ---
    for (int rr = 0; rr < 2; ++rr) {
      const int r = wave * 2 + rr;
      float mx = -3.0e38f;
      for (int c0 = lane; c0 < 512; c0 += 32) mx = fmaxf(mx, sc[r][c0]);
      for (int off = 16; off > 0; off >>= 1) mx = fmaxf(mx, __shfl_xor(mx, off, 32));
      const float oldm = rowmax[r];
      const float newm = fmaxf(oldm, mx);
      const float corr = __expf(oldm - newm);
      float lsum = 0.f;
      for (int c0 = lane; c0 < 512; c0 += 32) {
        float p = __expf(sc[r][c0] - newm);
        Pl[r][c0] = (_Float16)p;
        lsum += p;
      }
      for (int off = 16; off > 0; off >>= 1) lsum += __shfl_xor(lsum, off, 32);
      if (lane == 0) { rowmax[r] = newm; rowsum[r] = rowsum[r] * corr + lsum; rowscale[r] = corr; }
    }
    __syncthreads();

    // --- rescale running ctx, then ctx += P @ V_chunk ---
#pragma unroll
    for (int v = 0; v < 8; ++v) ctx[v] *= rowscale[rb + v];
    for (int kk = 0; kk < 16; ++kk) {
      v16h af, bf;
#pragma unroll
      for (int e = 0; e < 8; ++e) {
        af[e]     = Pl[arow][kk * 32 + kb + e];
        af[e + 8] = Pl[arow][kk * 32 + kb + 16 + e];
      }
      const _Float16* vrow = V + (bofs + ch * 512 + kk * 32) * HH + hofs + wave * 16 + arow;
#pragma unroll
      for (int e = 0; e < 8; ++e) {
        bf[e]     = vrow[(size_t)(kb + e) * HH];
        bf[e + 8] = vrow[(size_t)(kb + 16 + e) * HH];
      }
      ctx = wmma_f16(af, bf, ctx);
    }
    __syncthreads();
  }

  const int n = lane & 15;
#pragma unroll
  for (int v = 0; v < 8; ++v) {
    int r = rb + v;
    float val = ctx[v] / rowsum[r];
    Ctx[(bofs + qt * 16 + r) * HH + hofs + wave * 16 + n] = (_Float16)val;
  }
}

// ---------------- router tail ----------------
__global__ __launch_bounds__(256) void router_kernel(
    const _Float16* __restrict__ R1, const float* __restrict__ Wr2,
    const float* __restrict__ br2, float* __restrict__ disp,
    float* __restrict__ comb, float* __restrict__ probs,
    float* __restrict__ psum) {
  __shared__ float w2[HH * EE];
  for (int i = threadIdx.x; i < HH * EE; i += blockDim.x) w2[i] = Wr2[i];
  __syncthreads();

  const int m = blockIdx.x * blockDim.x + threadIdx.x;
  if (m >= MTOT) return;

  float lg[EE];
#pragma unroll
  for (int e = 0; e < EE; ++e) lg[e] = br2[e];
  const _Float16* row = R1 + (size_t)m * HH;
  for (int k = 0; k < HH; ++k) {
    float r = (float)row[k];
#pragma unroll
    for (int e = 0; e < EE; ++e) lg[e] += r * w2[k * EE + e];
  }
  float mx = lg[0];
#pragma unroll
  for (int e = 1; e < EE; ++e) mx = fmaxf(mx, lg[e]);
  float sum = 0.f, p[EE];
#pragma unroll
  for (int e = 0; e < EE; ++e) { p[e] = __expf(lg[e] - mx); sum += p[e]; }
  float inv = 1.f / sum;
#pragma unroll
  for (int e = 0; e < EE; ++e) {
    p[e] *= inv;
    probs[(size_t)m * EE + e] = p[e];
    atomicAdd(&psum[e], p[e]);
  }
  int i1 = 0; float v1 = p[0];
#pragma unroll
  for (int e = 1; e < EE; ++e) if (p[e] > v1) { v1 = p[e]; i1 = e; }
  int i2 = (i1 == 0) ? 1 : 0; float v2 = p[i2];
#pragma unroll
  for (int e = 0; e < EE; ++e) if (e != i1 && p[e] > v2) { v2 = p[e]; i2 = e; }
  float wn = 1.f / (v1 + v2);
  size_t base = (size_t)m * EE * CAPP;
  disp[base + (size_t)i1 * CAPP] = 1.f;
  disp[base + (size_t)i2 * CAPP] = 1.f;
  comb[base + (size_t)i1 * CAPP] = v1 * wn;
  comb[base + (size_t)i2 * CAPP] = v2 * wn;
}

__global__ void aux_kernel(const float* __restrict__ psum, float* __restrict__ out_aux) {
  __shared__ float sh[EE];
  int t = threadIdx.x;
  if (t < EE) {
    float pm = psum[t] * (1.f / (float)MTOT);
    sh[t] = pm * __logf(pm * (float)EE + 1e-9f);
  }
  __syncthreads();
  if (t == 0) {
    float s = 0.f;
    for (int e = 0; e < EE; ++e) s += sh[e];
    *out_aux = s;
  }
}

// ---------------- host launcher ----------------
extern "C" void kernel_launch(void* const* d_in, const int* in_sizes, int n_in,
                              void* d_out, int out_size, void* d_ws, size_t ws_size,
                              hipStream_t stream) {
  (void)in_sizes; (void)n_in; (void)out_size; (void)ws_size;

  const float* hs  = (const float*)d_in[0];
  const float* pos = (const float*)d_in[1];
  const float* We1 = (const float*)d_in[2];
  const float* be1 = (const float*)d_in[3];
  const float* We2 = (const float*)d_in[4];
  const float* be2 = (const float*)d_in[5];
  const float* Ws1 = (const float*)d_in[6];
  const float* bs1 = (const float*)d_in[7];
  const float* Ws2 = (const float*)d_in[8];
  const float* bs2 = (const float*)d_in[9];
  const float* Wq  = (const float*)d_in[10];
  const float* bq  = (const float*)d_in[11];
  const float* Wk  = (const float*)d_in[12];
  const float* bk  = (const float*)d_in[13];
  const float* Wv  = (const float*)d_in[14];
  const float* bv  = (const float*)d_in[15];
  const float* Wo  = (const float*)d_in[16];
  const float* bo  = (const float*)d_in[17];
  const float* Wr1 = (const float*)d_in[18];
  const float* br1 = (const float*)d_in[19];
  const float* Wr2 = (const float*)d_in[20];
  const float* br2 = (const float*)d_in[21];

  char* ws = (char*)d_ws;
  _Float16* w16  = (_Float16*)(ws + WS_W16);
  _Float16* we1p = (_Float16*)(ws + WS_WE1P);
  float*    seas = (float*)   (ws + WS_SEAS);
  float*    psum = (float*)   (ws + WS_PSUM);
  _Float16* bufA = (_Float16*)(ws + WS_ACT + 0 * ACT_STRIDE); // A0 -> ctx
  _Float16* bufB = (_Float16*)(ws + WS_ACT + 1 * ACT_STRIDE); // h1 -> xo
  _Float16* bufC = (_Float16*)(ws + WS_ACT + 2 * ACT_STRIDE); // x  -> r1
  _Float16* bufQ = (_Float16*)(ws + WS_ACT + 3 * ACT_STRIDE);
  _Float16* bufK = (_Float16*)(ws + WS_ACT + 4 * ACT_STRIDE); // Kt transposed
  _Float16* bufV = (_Float16*)(ws + WS_ACT + 5 * ACT_STRIDE);

  _Float16* fWe2 = w16 + 0 * W16_ELEMS;
  _Float16* fWq  = w16 + 1 * W16_ELEMS;
  _Float16* fWk  = w16 + 2 * W16_ELEMS;
  _Float16* fWv  = w16 + 3 * W16_ELEMS;
  _Float16* fWo  = w16 + 4 * W16_ELEMS;
  _Float16* fWr1 = w16 + 5 * W16_ELEMS;

  float* out_disp = (float*)d_out;
  float* out_comb = out_disp + DISP_ELEMS;
  float* out_prob = out_comb + DISP_ELEMS;
  float* out_aux  = out_prob + (size_t)MTOT * EE;

  {
    size_t n4 = (2 * DISP_ELEMS) / 4;
    zero_out_kernel<<<16384, 256, 0, stream>>>((float4*)d_out, n4, psum);
  }
  {
    int n = (int)W16_ELEMS, blk = (n + 255) / 256;
    cast_f16_kernel<<<blk, 256, 0, stream>>>(We2, fWe2, n);
    cast_f16_kernel<<<blk, 256, 0, stream>>>(Wq,  fWq,  n);
    cast_f16_kernel<<<blk, 256, 0, stream>>>(Wk,  fWk,  n);
    cast_f16_kernel<<<blk, 256, 0, stream>>>(Wv,  fWv,  n);
    cast_f16_kernel<<<blk, 256, 0, stream>>>(Wo,  fWo,  n);
    cast_f16_kernel<<<blk, 256, 0, stream>>>(Wr1, fWr1, n);
    pack_we1_kernel<<<(KPAD * HH + 255) / 256, 256, 0, stream>>>(We1, we1p);
  }
  build_a0_kernel<<<(int)(((size_t)MTOT * KPAD + 255) / 256), 256, 0, stream>>>(hs, bufA);
  seas_kernel<<<(SS * HQQ + 255) / 256, 256, 0, stream>>>(Ws1, bs1, Ws2, bs2, seas);

  dim3 g1(HH / 128, MTOT / 128);   // (8, 32)
  gemm16_kernel<<<g1, 256, 0, stream>>>(bufA, we1p, be1, bufB, MTOT, HH, KPAD, EPI_RELU, nullptr, nullptr);
  gemm16_kernel<<<g1, 256, 0, stream>>>(bufB, fWe2, be2, bufC, MTOT, HH, HH, EPI_POSSEAS, pos, seas);
  gemm16_kernel<<<g1, 256, 0, stream>>>(bufC, fWq, bq, bufQ, MTOT, HH, HH, EPI_NONE, nullptr, nullptr);
  gemm16_kernel<<<g1, 256, 0, stream>>>(bufC, fWk, bk, bufK, MTOT, HH, HH, EPI_TRANSK, nullptr, nullptr);
  gemm16_kernel<<<g1, 256, 0, stream>>>(bufC, fWv, bv, bufV, MTOT, HH, HH, EPI_NONE, nullptr, nullptr);
  attn_kernel<<<BB * NHH * (SS / 16), 256, 0, stream>>>(bufQ, bufK, bufV, bufA);
  gemm16_kernel<<<g1, 256, 0, stream>>>(bufA, fWo, bo, bufB, MTOT, HH, HH, EPI_NONE, nullptr, nullptr);
  gemm16_kernel<<<g1, 256, 0, stream>>>(bufB, fWr1, br1, bufC, MTOT, HH, HH, EPI_RELU, nullptr, nullptr);
  router_kernel<<<MTOT / 256, 256, 0, stream>>>(bufC, Wr2, br2, out_disp, out_comb, out_prob, psum);
  aux_kernel<<<1, 32, 0, stream>>>(psum, out_aux);
}
